// Attention_22771916603605
// MI455X (gfx1250) — compile-verified, matrix operand order checked by reading, not run on hardware
//
#include <hip/hip_runtime.h>

typedef float  v8f   __attribute__((ext_vector_type(8)));
typedef float  v2f   __attribute__((ext_vector_type(2)));
typedef __bf16 v16bf __attribute__((ext_vector_type(16)));
typedef __bf16 bf16_t;
typedef unsigned int u32x4 __attribute__((ext_vector_type(4)));
typedef int          i32x8 __attribute__((ext_vector_type(8)));
typedef int          i32x4 __attribute__((ext_vector_type(4)));

#define N_SEQ   4096
#define D_DIM   64
#define R_DIM   256
#define CHUNK   64
#define NCHUNK  64
#define NORMALIZER 0.35355339059327373f   /* 64^-0.25 */
#define RATIO      0.0625f                /* 256^-0.5 */
#define KEPS       1.0e-4f
#define AEPS       1.0e-6f

// LDS byte offsets inside the attention kernel's dynamic LDS block
#define QP_OFF   0        // bf16 [64][264]
#define KP_OFF   33792    // bf16 [64][264]
#define KPT_OFF  67584    // bf16 [256][72]
#define VT_OFF   104448   // bf16 [80][72]
#define CTXT_OFF 115968   // bf16 [80][264]
#define SM_OFF   158208   // bf16 [64][72]
#define O_OFF    167424   // f32  [64][84]
#define DINV_OFF 188928   // f32  [64]

// ---------------------------------------------------------------------------
// bf16 A/B fragment loader (wave32). Storage is [row][k] with k contiguous, so
// this serves A operands directly and B operands from a k-major (transposed)
// copy. Layout per ISA 7.12.2: lane L holds row L%16; VGPR v (pair 2v,2v+1)
// holds K = k0 + {half*8 + 2v}           for v<4,
//            k0 + {16 + half*8 + 2(v-4)} for v>=4,  half = L/16.
// ---------------------------------------------------------------------------
__device__ __forceinline__ v16bf ld_frag16(const bf16_t* base, int stride,
                                           int row0, int k0, int lane) {
  const int m  = row0 + (lane & 15);
  const int hl = lane >> 4;
  const bf16_t* rp = base + m * stride + k0;
  v16bf f;
#pragma unroll
  for (int i = 0; i < 8; ++i) {
    const int k = (i < 4) ? (hl * 8 + i * 2) : (16 + hl * 8 + (i - 4) * 2);
    f[2 * i]     = rp[k];
    f[2 * i + 1] = rp[k + 1];
  }
  return f;
}

// ---------------------------------------------------------------------------
// TDM: DMA a [rows x 256] bf16 tile (contiguous rows of 512 B) from global
// into LDS at lds_off, inserting 16 B of padding after every 512 B so rows
// land with a 264-element LDS stride.  D# layout per ISA 8.3/8.4:
//   g0: count=1 | lds_addr | global_addr | type=2
//   g1: data_size=2B, pad_enable, pad_interval=128 dwords, pad_amount=4 dwords,
//       tensor_dim0=256, tensor_dim1=rows, tile_dim0=256, tile_dim1=rows,
//       tensor_dim0_stride=256
// Toolchain here is clang-23 (therock): 6-arg builtin form.
// ---------------------------------------------------------------------------
#if __has_builtin(__builtin_amdgcn_tensor_load_to_lds)
#define HAVE_TDM 1
__device__ __forceinline__ void tdm_load_rows(const void* gaddr,
                                              unsigned int lds_off, int rows) {
  const unsigned long long ga = (unsigned long long)gaddr;
  u32x4 g0;
  g0[0] = 1u;                                        // count=1, user descriptor
  g0[1] = lds_off;                                   // lds_addr (bytes)
  g0[2] = (unsigned int)(ga & 0xffffffffull);        // global_addr[31:0]
  g0[3] = (unsigned int)((ga >> 32) & 0x1ffffffu) | (2u << 30);  // addr hi | type=2
  i32x8 g1;
  g1[0] = (1 << 16) | (1 << 20) | (6 << 22) | (3 << 25);  // 2B elems, pad 4dw/128dw
  g1[1] = (int)(256u << 16);                 // tensor_dim0[15:0] in [31:16]
  g1[2] = (int)((unsigned)rows << 16);       // dim0 hi=0 | tensor_dim1[15:0]
  g1[3] = (int)(256u << 16);                 // dim1 hi=0 | tile_dim0=256
  g1[4] = rows;                              // tile_dim1=rows, tile_dim2=0
  g1[5] = 256;                               // tensor_dim0_stride lo
  g1[6] = 0;                                 // stride hi | dim1_stride lo
  g1[7] = 0;
  i32x4 z4 = {0, 0, 0, 0};
  i32x8 z8 = {0, 0, 0, 0, 0, 0, 0, 0};
  __builtin_amdgcn_tensor_load_to_lds(g0, g1, z4, z4, z8, 0);
}
#else
#define HAVE_TDM 0
#endif

// ---------------------------------------------------------------------------
// Feature-map kernel: dash = (x * d^-0.25) @ P^T via V_WMMA_F32_16X16X4_F32.
// mode 0: keys pass 1 -> per-block max of dash only
// mode 1: queries     -> row-wise stab, write bf16 features
// mode 2: keys pass 2 -> global stab from gmax_ptr, write bf16 features
// grid = 1024 blocks (16 bh * 64 row-tiles), 256 threads (8 waves)
// ---------------------------------------------------------------------------
__global__ void performer_proj_kernel(const float* __restrict__ data,
                                      const float* __restrict__ proj,
                                      bf16_t* __restrict__ feat,
                                      float* __restrict__ blockmax,
                                      const float* __restrict__ gmax_ptr,
                                      int mode) {
  extern __shared__ char smem[];
  float* p_lds = (float*)(smem + 0);       // [256][65]
  float* q_lds = (float*)(smem + 66560);   // [64][65]
  float* dash  = (float*)(smem + 83200);   // [64][257]
  float* diag  = (float*)(smem + 148992);  // [64]
  float* stab  = (float*)(smem + 149248);  // [64]
  float* red   = (float*)(smem + 149504);  // [256]

  const int t    = threadIdx.x;
  const int lane = t & 31;
  const int wave = t >> 5;
  const int l16  = lane & 15;
  const int hl   = lane >> 4;
  const long rowbase = (long)(blockIdx.x >> 6) * N_SEQ + (long)(blockIdx.x & 63) * 64;

  for (int i = t; i < R_DIM * D_DIM; i += 256) {
    const int r = i >> 6, c = i & 63;
    p_lds[r * 65 + c] = proj[i];
  }
  for (int i = t; i < 64 * D_DIM; i += 256) {
    const int r = i >> 6, c = i & 63;
    q_lds[r * 65 + c] = data[(rowbase + r) * D_DIM + c] * NORMALIZER;
  }
  __syncthreads();

  if (t < 64) {                       // diag = 0.5 * norm^2 * sum(x^2)
    float s = 0.f;
#pragma unroll
    for (int c = 0; c < 64; ++c) { const float x = q_lds[t * 65 + c]; s += x * x; }
    diag[t] = 0.5f * s;
  }

  // 4x16 = 64 output tiles of 16x16, 8 per wave, K = 64 -> 16 WMMA f32 each
  for (int tt = wave * 8; tt < wave * 8 + 8; ++tt) {
    const int m0 = (tt >> 4) * 16;
    const int n0 = (tt & 15) * 16;
    v8f acc = {};
#pragma unroll
    for (int kk = 0; kk < 16; ++kk) {
      const int k0 = kk * 4 + hl * 2;   // f32 A/B: VGPR v holds K = half*2 + v
      v2f a, b;
      { const float* rp = q_lds + (m0 + l16) * 65 + k0; a[0] = rp[0]; a[1] = rp[1]; }
      { const float* rp = p_lds + (n0 + l16) * 65 + k0; b[0] = rp[0]; b[1] = rp[1]; }
      acc = __builtin_amdgcn_wmma_f32_16x16x4_f32(false, a, false, b,
                                                  (short)0, acc, false, false);
    }
    const int n = n0 + l16;
#pragma unroll
    for (int i = 0; i < 8; ++i) dash[(m0 + i + hl * 8) * 257 + n] = acc[i];
  }
  __syncthreads();

  if (mode == 0) {
    float m = -3.4e38f;
    for (int j = 0; j < 64; ++j) {
      const int idx = t * 64 + j;
      m = fmaxf(m, dash[(idx >> 8) * 257 + (idx & 255)]);
    }
    red[t] = m;
    __syncthreads();
    for (int s = 128; s > 0; s >>= 1) {
      if (t < s) red[t] = fmaxf(red[t], red[t + s]);
      __syncthreads();
    }
    if (t == 0) blockmax[blockIdx.x] = red[0];
  } else {
    if (mode == 1) {
      if (t < 64) {
        float m = -3.4e38f;
        for (int r = 0; r < R_DIM; ++r) m = fmaxf(m, dash[t * 257 + r]);
        stab[t] = m;
      }
      __syncthreads();
    }
    const float g = (mode == 2) ? gmax_ptr[0] : 0.f;
    for (int j = 0; j < 64; ++j) {         // row j, col t (coalesced)
      const float s   = (mode == 1) ? stab[j] : g;
      const float val = RATIO * (__expf(dash[j * 257 + t] - diag[j] - s) + KEPS);
      feat[(rowbase + j) * R_DIM + t] = (bf16_t)val;
    }
  }
}

__global__ void performer_reduce_max_kernel(const float* __restrict__ in,
                                            float* __restrict__ out, int n) {
  __shared__ float red[256];
  float m = -3.4e38f;
  for (int i = threadIdx.x; i < n; i += 256) m = fmaxf(m, in[i]);
  red[threadIdx.x] = m;
  __syncthreads();
  for (int s = 128; s > 0; s >>= 1) {
    if (threadIdx.x < s) red[threadIdx.x] = fmaxf(red[threadIdx.x], red[threadIdx.x + s]);
    __syncthreads();
  }
  if (threadIdx.x == 0) *out = red[0];
}

// ---------------------------------------------------------------------------
// Chunked causal linear attention. grid = 16 blocks (b*h), 256 threads.
// State ctx_ext [256 r][80] lives in chained WMMA accumulators (10 tiles/wave):
//   cols 0..63 = ctx, col 64 = k_cum (via v-ones column), col 65 = const 1.
// Per chunk: S = qp kp^T (masked tril) ; O = qp ctx_prev + tril(S) v_ext ;
//   D_i = O[i][64] + 1e-6*O[i][65] ; out = O[:, :64] / D ; ctx += kp^T v_ext.
// qp/kp chunk tiles arrive via the Tensor Data Mover (padded-row DMA to LDS).
// ---------------------------------------------------------------------------
__global__ void performer_attn_kernel(const bf16_t* __restrict__ qp,
                                      const bf16_t* __restrict__ kp,
                                      const float* __restrict__ vv,
                                      float* __restrict__ out) {
  extern __shared__ char smem[];
  bf16_t* qp_lds  = (bf16_t*)(smem + QP_OFF);    // [64][264]
  bf16_t* kp_lds  = (bf16_t*)(smem + KP_OFF);    // [64][264]  (row j, k-major in r)
  bf16_t* kpT_lds = (bf16_t*)(smem + KPT_OFF);   // [256][72]  (row r, k-major in j)
  bf16_t* vT_lds  = (bf16_t*)(smem + VT_OFF);    // [80][72]   (row e_ext, k-major in j)
  bf16_t* ctxT    = (bf16_t*)(smem + CTXT_OFF);  // [80][264]  (row e_ext, k-major in r)
  bf16_t* smask   = (bf16_t*)(smem + SM_OFF);    // [64][72]
  float*  o_lds   = (float*)(smem + O_OFF);      // [64][84]
  float*  dinv    = (float*)(smem + DINV_OFF);   // [64]

  const int t    = threadIdx.x;
  const int lane = t & 31;
  const int hl   = lane >> 4;
  const int l16  = lane & 15;
  const int wave = t >> 5;

  v8f ctx_acc[10];
#pragma unroll
  for (int tl = 0; tl < 10; ++tl) {
    const int gt = wave * 10 + tl;
    const float init = (((gt % 5) * 16 + l16) == 65) ? 1.0f : 0.0f;  // const-1 col
    v8f z;
#pragma unroll
    for (int i = 0; i < 8; ++i) z[i] = init;
    ctx_acc[tl] = z;
  }

  for (int ch = 0; ch < NCHUNK; ++ch) {
    const long seq0 = (long)blockIdx.x * N_SEQ + (long)ch * CHUNK;
    const bf16_t* qg = qp + seq0 * R_DIM;
    const bf16_t* kg = kp + seq0 * R_DIM;
    const float*  vg = vv + seq0 * D_DIM;

    // -- phase 0: async-DMA qp/kp tiles, dump ctx regs, stage kp^T / v^T -----
#if HAVE_TDM
    if (wave == 0) {
      tdm_load_rows(qg, QP_OFF, CHUNK);   // [64][256] -> padded [64][264]
      tdm_load_rows(kg, KP_OFF, CHUNK);
    }
#else
    for (int j = 0; j < 8; ++j) {                   // 128-bit copies
      const int u = t + j * 256;                    // 0..2047
      const int row = u >> 5, c8 = (u & 31) * 8;
      *(uint4*)(qp_lds + row * 264 + c8) = *(const uint4*)(qg + row * R_DIM + c8);
      *(uint4*)(kp_lds + row * 264 + c8) = *(const uint4*)(kg + row * R_DIM + c8);
    }
#endif
#pragma unroll
    for (int tl = 0; tl < 10; ++tl) {               // ctx regs -> bf16 transposed
      const int gt = wave * 10 + tl;
      const int r0 = (gt / 5) * 16;
      const int n  = (gt % 5) * 16 + l16;
#pragma unroll
      for (int i = 0; i < 8; ++i)
        ctxT[n * 264 + r0 + i + hl * 8] = (bf16_t)ctx_acc[tl][i];
    }
    for (int j = 0; j < 64; ++j)                    // kp^T scatter (coalesced reads)
      kpT_lds[t * 72 + j] = kg[j * R_DIM + t];
    for (int j = 0; j < 16; ++j) {                  // v -> bf16 transposed
      const int u = t + j * 256;
      const int i = u >> 6, e = u & 63;
      vT_lds[e * 72 + i] = (bf16_t)vg[i * D_DIM + e];
    }
    for (int j = 0; j < 4; ++j) {                   // ones col 64, zeros 65..79
      const int u  = t + j * 256;
      const int er = 64 + (u >> 6), i = u & 63;
      vT_lds[er * 72 + i] = (er == 64) ? (bf16_t)1.0f : (bf16_t)0.0f;
    }
    if (ch + 1 < NCHUNK) {                          // prefetch next chunk
      const char* qn = (const char*)(qg + (long)CHUNK * R_DIM);
      const char* kn = (const char*)(kg + (long)CHUNK * R_DIM);
      const char* vn = (const char*)(vg + (long)CHUNK * D_DIM);
      __builtin_prefetch(qn + t * 128, 0, 1);
      __builtin_prefetch(kn + t * 128, 0, 1);
      __builtin_prefetch(vn + t * 64, 0, 1);
    }
#if HAVE_TDM
    __builtin_amdgcn_s_wait_tensorcnt(0);
#endif
    __syncthreads();

    // -- phase 1: S = qp @ kp^T, write masked tril as bf16 -------------------
    for (int s = wave * 2; s < wave * 2 + 2; ++s) {
      const int m0 = (s >> 2) * 16;
      const int n0 = (s & 3) * 16;
      v8f acc = {};
#pragma unroll
      for (int kk = 0; kk < 8; ++kk) {
        v16bf a = ld_frag16(qp_lds, 264, m0, kk * 32, lane);
        v16bf b = ld_frag16(kp_lds, 264, n0, kk * 32, lane);
        acc = __builtin_amdgcn_wmma_f32_16x16x32_bf16(false, a, false, b,
                                                      (short)0, acc, false, false);
      }
      const int n = n0 + l16;
#pragma unroll
      for (int i = 0; i < 8; ++i) {
        const int m = m0 + i + hl * 8;
        smask[m * 72 + n] = (n <= m) ? (bf16_t)acc[i] : (bf16_t)0.0f;
      }
    }
    __syncthreads();

    // -- phase 2: O = qp @ ctx_prev + tril(S) @ v_ext  (64 x 80) -------------
    for (int tt = wave; tt < 20; tt += 8) {
      const int m0 = (tt / 5) * 16;
      const int n0 = (tt % 5) * 16;
      v8f acc = {};
#pragma unroll
      for (int kk = 0; kk < 8; ++kk) {                    // K = 256 over r
        v16bf a = ld_frag16(qp_lds, 264, m0, kk * 32, lane);
        v16bf b = ld_frag16(ctxT,   264, n0, kk * 32, lane);
        acc = __builtin_amdgcn_wmma_f32_16x16x32_bf16(false, a, false, b,
                                                      (short)0, acc, false, false);
      }
#pragma unroll
      for (int kk = 0; kk < 2; ++kk) {                    // K = 64 over chunk
        v16bf a = ld_frag16(smask,  72, m0, kk * 32, lane);
        v16bf b = ld_frag16(vT_lds, 72, n0, kk * 32, lane);
        acc = __builtin_amdgcn_wmma_f32_16x16x32_bf16(false, a, false, b,
                                                      (short)0, acc, false, false);
      }
      const int n = n0 + l16;
#pragma unroll
      for (int i = 0; i < 8; ++i)
        o_lds[(m0 + i + hl * 8) * 84 + n] = acc[i];
    }
    __syncthreads();

    // -- phase 3: denominators ----------------------------------------------
    if (t < 64)
      dinv[t] = 1.0f / (o_lds[t * 84 + 64] + AEPS * o_lds[t * 84 + 65]);
    __syncthreads();

    // -- phase 4: scaled output ---------------------------------------------
    float* og = out + seq0 * D_DIM;
    for (int j = 0; j < 16; ++j) {
      const int u = t + j * 256;
      const int i = u >> 6, e = u & 63;
      og[i * D_DIM + e] = o_lds[i * 84 + e] * dinv[i];
    }

    // -- phase 5: ctx_ext += kp^T @ v_ext (chained accumulators) -------------
#pragma unroll
    for (int tl = 0; tl < 10; ++tl) {
      const int gt = wave * 10 + tl;
      const int r0 = (gt / 5) * 16;
      const int e0 = (gt % 5) * 16;
#pragma unroll
      for (int kk = 0; kk < 2; ++kk) {
        v16bf a = ld_frag16(kpT_lds, 72, r0, kk * 32, lane);
        v16bf b = ld_frag16(vT_lds,  72, e0, kk * 32, lane);
        ctx_acc[tl] = __builtin_amdgcn_wmma_f32_16x16x32_bf16(
            false, a, false, b, (short)0, ctx_acc[tl], false, false);
      }
    }
    __syncthreads();
  }
}

// ---------------------------------------------------------------------------
extern "C" void kernel_launch(void* const* d_in, const int* in_sizes, int n_in,
                              void* d_out, int out_size, void* d_ws, size_t ws_size,
                              hipStream_t stream) {
  (void)in_sizes; (void)n_in; (void)out_size; (void)ws_size;
  const float* q = (const float*)d_in[0];
  const float* k = (const float*)d_in[1];
  const float* v = (const float*)d_in[2];
  const float* P = (const float*)d_in[3];
  float* out = (float*)d_out;

  char* ws = (char*)d_ws;
  bf16_t* qp       = (bf16_t*)(ws);               // 16 * 4096 * 256 bf16 = 32 MiB
  bf16_t* kp       = (bf16_t*)(ws + 33554432);    // 32 MiB
  float*  blockmax = (float*)(ws + 67108864);     // 1024 f32
  float*  gmax     = (float*)(ws + 67112960);     // 1 f32

  constexpr size_t LDS_PROJ = 150528;
  constexpr size_t LDS_ATTN = 189184;

  performer_proj_kernel<<<dim3(1024), dim3(256), LDS_PROJ, stream>>>(
      q, P, qp, nullptr, nullptr, 1);
  performer_proj_kernel<<<dim3(1024), dim3(256), LDS_PROJ, stream>>>(
      k, P, nullptr, blockmax, nullptr, 0);
  performer_reduce_max_kernel<<<dim3(1), dim3(256), 0, stream>>>(blockmax, gmax, 1024);
  performer_proj_kernel<<<dim3(1024), dim3(256), LDS_PROJ, stream>>>(
      k, P, kp, nullptr, gmax, 2);
  performer_attn_kernel<<<dim3(16), dim3(256), LDS_ATTN, stream>>>(qp, kp, v, out);
}
